// EMRouting_73040213835986
// MI455X (gfx1250) — compile-verified, matrix operand order checked by reading, not run on hardware
//
#include <hip/hip_runtime.h>
#include <math.h>

// ---- problem constants (from reference) ----
#define B_ 2
#define H_ 12
#define W_ 12
#define K_ 3
#define I_ 32
#define O_ 32
#define P_ 16                    // a*a
#define C_ (K_*K_*I_)            // 288 input capsules per cell
#define HR_ (H_ + K_ - 1)        // 14
#define WR_ (W_ + K_ - 1)        // 14
#define NCELL (B_*H_*W_)         // 288
#define EPSF 1e-7f
#define MINVAR 0.0005f
#define TWO_PI 6.28318530717958647692f

typedef __attribute__((ext_vector_type(16))) _Float16 v16h;
typedef __attribute__((ext_vector_type(8)))  float    v8f;
typedef __attribute__((ext_vector_type(4)))  int      v4i;

#if __has_builtin(__builtin_amdgcn_global_load_async_to_lds_b128) && \
    __has_builtin(__builtin_amdgcn_s_wait_asynccnt)
#define HAVE_ASYNC_LDS 1
typedef __attribute__((address_space(1))) v4i GV4;   // global int4 (prints as __device__)
typedef __attribute__((address_space(3))) v4i LV4;   // LDS int4 (prints as __shared__)
#else
#define HAVE_ASYNC_LDS 0
#endif

// order-preserving float<->uint bijection for atomic max over mixed-sign floats
__device__ __forceinline__ unsigned encodeF(float f) {
  unsigned b = __float_as_uint(f);
  return (b & 0x80000000u) ? ~b : (b | 0x80000000u);
}
__device__ __forceinline__ float decodeF(unsigned u) {
  unsigned b = (u & 0x80000000u) ? (u & 0x7FFFFFFFu) : ~u;
  return __uint_as_float(b);
}

// ---------------------------------------------------------------------------
// m_step: one block per (cell, o). The 18KB votes tile (288 c x 16 p) is
// staged into LDS ONCE via the gfx1250 async global->LDS path (ASYNCcnt),
// then both the mu pass (WMMA ones-reduction on the matrix pipe) and the
// sigma pass read LDS -> votes touched once per m_step instead of twice.
//   mu: D += Ones(16x32 f16) x B(32x16 f16), B[k][n] = Ra[c0+k]*V[c0+k][n]
// row 0 of the f32 accumulator = Sum_c Ra*votes for all 16 pose components.
// ---------------------------------------------------------------------------
__global__ __launch_bounds__(256) void mstep_kernel(
    const float* __restrict__ votes, const float* __restrict__ acts,
    const float* __restrict__ Rbuf,
    const float* __restrict__ beta_a, const float* __restrict__ beta_u,
    float* __restrict__ muO, float* __restrict__ sigO, float* __restrict__ ajO,
    float inv_temp, int uniformR,
    float* __restrict__ outPoses, float* __restrict__ outActs)
{
  __shared__ float sV[C_ * P_];    // 18KB votes tile for this (cell, oo)
  __shared__ float sRa[C_];
  __shared__ float sRed[256];
  __shared__ float sPart[128];     // 8 waves x 16 partial column sums
  __shared__ float sMu[P_];
  __shared__ float sSig[P_];
  __shared__ float sSumR;

  const int blk  = blockIdx.x;     // cell*O + oo
  const int cell = blk / O_;
  const int oo   = blk % O_;
  const int t    = threadIdx.x;
  const int lane = t & 31;
  const int wave = t >> 5;

  const size_t cellC0 = (size_t)cell * C_;
  // element (c, p) of this (cell, oo): vbase[c*(O_*P_) + p]
  const float* vbase = votes + (cellC0 * O_ + (size_t)oo) * P_;

  // ---- stage votes tile: 288 rows x 64B, 1152 x b128 transfers ----
#if HAVE_ASYNC_LDS
  for (int idx = t; idx < C_ * 4; idx += 256) {
    const int c = idx >> 2, q = (idx & 3) * 4;
    __builtin_amdgcn_global_load_async_to_lds_b128(
        (GV4*)(vbase + (size_t)c * (O_ * P_) + q),
        (LV4*)(&sV[c * P_ + q]),
        /*imm offset*/0, /*cpol*/0);
  }
#else
  for (int idx = t; idx < C_ * 4; idx += 256) {
    const int c = idx >> 2, q = (idx & 3) * 4;
    const float4 vv = *(const float4*)(vbase + (size_t)c * (O_ * P_) + q);
    *(float4*)(&sV[c * P_ + q]) = vv;
  }
#endif

  // ---- Ra[c] = R[c,oo] * acts[c] while the copies are in flight ----
  for (int c = t; c < C_; c += 256) {
    float a = acts[cellC0 + c];
    float r = uniformR ? (1.0f / (float)O_) : Rbuf[(cellC0 + c) * O_ + oo];
    sRa[c] = r * a;
  }
#if HAVE_ASYNC_LDS
  __builtin_amdgcn_s_wait_asynccnt(0);
#endif
  __syncthreads();

  // ---- sumR = Sum_c Ra[c] ----
  float x = 0.f;
  for (int c = t; c < C_; c += 256) x += sRa[c];
  sRed[t] = x;
  __syncthreads();
  for (int off = 128; off > 0; off >>= 1) {
    if (t < off) sRed[t] += sRed[t + off];
    __syncthreads();
  }
  if (t == 0) sSumR = sRed[0];
  __syncthreads();
  const float sumR = sSumR;
  const float invS = 1.0f / (sumR + EPSF);

  // ---- mu numerators via WMMA ones-reduction over c (B built from LDS) ----
  v16h aOnes;
  #pragma unroll
  for (int j = 0; j < 16; ++j) aOnes[j] = (_Float16)1.0f;

  v8f acc = {};
  const int p16   = lane & 15;
  const int kbase = (lane < 16) ? 0 : 16;    // B 32x16 f16: lanes 0-15 K=0..15, 16-31 K=16..31
  for (int ch = wave; ch < 9; ch += 8) {     // 9 chunks of 32 c; wave-uniform loop (EXEC all 1s)
    const int c0 = ch * 32;
    v16h bv;
    #pragma unroll
    for (int j = 0; j < 16; ++j) {
      const int c = c0 + kbase + j;
      bv[j] = (_Float16)(sRa[c] * sV[c * P_ + p16]);
    }
    acc = __builtin_amdgcn_wmma_f32_16x16x32_f16(
        /*neg_a=*/false, aOnes, /*neg_b=*/false, bv,
        /*c_mod=*/(short)0, acc, /*reuse_a=*/false, /*reuse_b=*/false);
  }
  if (lane < 16) sPart[wave * 16 + lane] = acc[0];  // D row M=0: (M=0, N=lane)
  __syncthreads();
  if (t < 16) {
    float m = 0.f;
    #pragma unroll
    for (int wv = 0; wv < 8; ++wv) m += sPart[wv * 16 + t];
    sMu[t] = m * invS;
  }
  __syncthreads();

  // ---- sigma from the LDS tile: Sum_c Ra*(v-mu)^2 / (sumR+eps) + MIN_VAR_ADD ----
  {
    const int pp    = t & 15;        // pose component
    const int slice = t >> 4;        // 16 slices x 18 c
    const float mup = sMu[pp];
    float sg = 0.f;
    #pragma unroll 6
    for (int q = 0; q < 18; ++q) {
      const int c = slice * 18 + q;
      float d = sV[c * P_ + pp] - mup;
      sg += sRa[c] * d * d;
    }
    sRed[t] = sg;
  }
  __syncthreads();
  if (t < 16) {
    float s2 = 0.f;
    #pragma unroll
    for (int sl = 0; sl < 16; ++sl) s2 += sRed[sl * 16 + t];
    sSig[t] = s2 * invS + MINVAR;
  }
  __syncthreads();

  // ---- cost -> a_j ----
  if (t < 16) sRed[t] = (beta_u[oo] - 0.5f * logf(sSig[t] + EPSF)) * sumR;  // MEAN_DATA==1
  __syncthreads();
  if (t == 0) {
    float cs = 0.f;
    #pragma unroll
    for (int p = 0; p < 16; ++p) cs += sRed[p];
    float z  = inv_temp * (beta_a[oo] - cs);
    float aj = 1.0f / (1.0f + expf(-z));
    ajO[cell * O_ + oo] = aj;
    if (outActs) outActs[cell * O_ + oo] = aj;
  }
  if (t < 16) {
    muO [(cell * O_ + oo) * P_ + t] = sMu[t];
    sigO[(cell * O_ + oo) * P_ + t] = sSig[t];
    if (outPoses) outPoses[(cell * O_ + oo) * P_ + t] = sMu[t];
  }
}

// init the global-max accumulator (encoded: 0 is below every encoded float)
__global__ void init_kernel(unsigned* __restrict__ maxbits) {
  if (blockIdx.x == 0 && threadIdx.x == 0) *maxbits = 0u;
}

// ---------------------------------------------------------------------------
// e1: lognum[(cell,c,o)] = log(a_j) - Sum_p [ log(2*pi*sig) + (v-mu)^2/(2 sig) ]
//     + global max via encoded atomicMax (one atomic per block)
// ---------------------------------------------------------------------------
__global__ __launch_bounds__(256) void e1_kernel(
    const float* __restrict__ votes, const float* __restrict__ mu,
    const float* __restrict__ sig, const float* __restrict__ aj,
    float* __restrict__ lognum, unsigned* __restrict__ maxbits)
{
  const int co    = blockIdx.x * 256 + threadIdx.x;  // (cellC, oo); grid sized exactly
  const int oo    = co & 31;
  const int cellC = co >> 5;
  const int cell  = cellC / C_;

  const float* v = votes + (size_t)co * P_;
  const float* m = mu  + ((size_t)cell * O_ + oo) * P_;
  const float* s = sig + ((size_t)cell * O_ + oo) * P_;

  float lp = 0.f;
  #pragma unroll
  for (int p = 0; p < 16; ++p) {
    float sp = s[p];
    float d  = v[p] - m[p];
    lp += logf(TWO_PI * sp) + d * d / (2.0f * sp);
  }
  float ln = logf(aj[cell * O_ + oo]) - lp;
  lognum[co] = ln;

  __shared__ float sm[256];
  sm[threadIdx.x] = ln;
  __syncthreads();
  for (int off = 128; off > 0; off >>= 1) {
    if (threadIdx.x < off) sm[threadIdx.x] = fmaxf(sm[threadIdx.x], sm[threadIdx.x + off]);
    __syncthreads();
  }
  if (threadIdx.x == 0) atomicMax(maxbits, encodeF(sm[0]));
}

// ---------------------------------------------------------------------------
// e2: dense[Y,X,b,i] = Sum over contributing (y,x,kh,kw) and o of exp(lognum-M)
//     gather formulation -> deterministic, no float atomics
// ---------------------------------------------------------------------------
__global__ __launch_bounds__(256) void e2_kernel(
    const float* __restrict__ lognum, const unsigned* __restrict__ maxbits,
    float* __restrict__ dense)
{
  const int tid = blockIdx.x * 256 + threadIdx.x;   // HR_*WR_*B_*I_ = 12544
  if (tid >= HR_ * WR_ * B_ * I_) return;
  const int ii = tid % I_;
  const int bb = (tid / I_) % B_;
  const int X  = (tid / (I_ * B_)) % WR_;
  const int Y  =  tid / (I_ * B_ * WR_);
  const float M = decodeF(*maxbits);

  float s = 0.f;
  for (int kh = 0; kh < K_; ++kh) {
    const int y = Y - kh;
    if ((unsigned)y >= (unsigned)H_) continue;
    for (int kw = 0; kw < K_; ++kw) {
      const int x = X - kw;
      if ((unsigned)x >= (unsigned)W_) continue;
      const size_t cellC = ((((size_t)(bb * H_ + y) * W_ + x) * K_ + kh) * K_ + kw) * I_ + ii;
      const float* ln = lognum + cellC * O_;
      #pragma unroll
      for (int oo = 0; oo < O_; ++oo) s += expf(ln[oo] - M);
    }
  }
  dense[tid] = s;
}

// ---------------------------------------------------------------------------
// e3: R[(cell,c,o)] = exp(lognum - M) / (dense[y+kh, x+kw, b, i] + eps)
// ---------------------------------------------------------------------------
__global__ __launch_bounds__(256) void e3_kernel(
    const float* __restrict__ lognum, const unsigned* __restrict__ maxbits,
    const float* __restrict__ dense, float* __restrict__ Rbuf)
{
  const int tid = blockIdx.x * 256 + threadIdx.x;  // cellC in [0, NCELL*C_); grid exact
  int r = tid;
  const int ii = r % I_; r /= I_;
  const int kw = r % K_; r /= K_;
  const int kh = r % K_; r /= K_;
  const int xw = r % W_; r /= W_;
  const int yh = r % H_;
  const int bb = r / H_;
  const int didx = ((yh + kh) * WR_ + (xw + kw)) * (B_ * I_) + bb * I_ + ii;

  const float M    = decodeF(*maxbits);
  const float invD = 1.0f / (dense[didx] + EPSF);
  const float* ln  = lognum + (size_t)tid * O_;
  float*       Ro  = Rbuf   + (size_t)tid * O_;
  #pragma unroll
  for (int oo = 0; oo < O_; ++oo) Ro[oo] = expf(ln[oo] - M) * invD;
}

// ---------------------------------------------------------------------------
extern "C" void kernel_launch(void* const* d_in, const int* in_sizes, int n_in,
                              void* d_out, int out_size, void* d_ws, size_t ws_size,
                              hipStream_t stream) {
  const float* votes  = (const float*)d_in[0];
  const float* acts   = (const float*)d_in[1];
  const float* beta_a = (const float*)d_in[2];
  const float* beta_u = (const float*)d_in[3];

  float* out     = (float*)d_out;
  float* poses   = out;                              // NCELL*O_*P_ = 147456
  float* actsOut = out + (size_t)NCELL * O_ * P_;    // NCELL*O_    = 9216

  // workspace carve-up (~22.6 MB)
  float* ws     = (float*)d_ws;
  float* Rbuf   = ws;                                         // NCELL*C_*O_ = 2,654,208
  float* lognum = Rbuf   + (size_t)NCELL * C_ * O_;           // 2,654,208
  float* mu     = lognum + (size_t)NCELL * C_ * O_;           // 147,456
  float* sig    = mu     + (size_t)NCELL * O_ * P_;           // 147,456
  float* aj     = sig    + (size_t)NCELL * O_ * P_;           // 9,216
  float* dense  = aj     + (size_t)NCELL * O_;                // 12,544
  unsigned* maxbits = (unsigned*)(dense + HR_ * WR_ * B_ * I_);

  // inv_temp = FINAL_LAMBDA * (1 - 0.95^(it+1)); m_step calls at it=0 and it=ITERATIONS=2
  const float it0 = 0.01f * (1.0f - 0.95f);
  const float itF = 0.01f * (1.0f - 0.95f * 0.95f * 0.95f);

  // 1) m_step with uniform R (no R init needed)
  mstep_kernel<<<NCELL * O_, 256, 0, stream>>>(votes, acts, Rbuf, beta_a, beta_u,
                                               mu, sig, aj, it0, /*uniformR=*/1,
                                               nullptr, nullptr);
  // 2) e_step
  init_kernel<<<1, 64, 0, stream>>>(maxbits);
  e1_kernel<<<(NCELL * C_ * O_) / 256, 256, 0, stream>>>(votes, mu, sig, aj, lognum, maxbits);
  e2_kernel<<<(HR_ * WR_ * B_ * I_ + 255) / 256, 256, 0, stream>>>(lognum, maxbits, dense);
  e3_kernel<<<(NCELL * C_) / 256, 256, 0, stream>>>(lognum, maxbits, dense, Rbuf);
  // 3) final m_step writes outputs
  mstep_kernel<<<NCELL * O_, 256, 0, stream>>>(votes, acts, Rbuf, beta_a, beta_u,
                                               mu, sig, aj, itF, /*uniformR=*/0,
                                               poses, actsOut);
}
